// CPABTransform2D_BIS_22952305229998
// MI455X (gfx1250) — matrix operand order repaired and checked by reading, not compiled, not run on hardware
//
#include <hip/hip_runtime.h>

typedef __attribute__((ext_vector_type(16))) _Float16 v16h;
typedef __attribute__((ext_vector_type(8)))  float    v8f;

#define EPS_CLIP   1e-7f
#define MLP_HALVES 10240     // W1 tiles(4096) + W2 tiles(4096) + Cw tiles(2048)
#define COEF_STR   33        // 32 coefs + 1 pad (bank-conflict-free)

union H2 { unsigned u; _Float16 h[2]; };

// A-matrix 16-bit K index for (lane, half): contiguous K = 16*(lane>>4) + h (matches B fills)
__device__ __forceinline__ int kkA(int lane, int h) {
    return h + 16 * (lane >> 4);
}

// ---------------------------------------------------------------------------
// Prep: build f16 weight tiles (A-layout) + fused Cw = basis @ wout, Cb = basis @ bout
// mlp slot 0 = params2 (runs first), slot 1 = params1.
// ---------------------------------------------------------------------------
__global__ void prep_weights(
    const float* __restrict__ w1_a, const float* __restrict__ w2_a,
    const float* __restrict__ wout_a, const float* __restrict__ bout_a,
    const float* __restrict__ w1_b, const float* __restrict__ w2_b,
    const float* __restrict__ wout_b, const float* __restrict__ bout_b,
    const float* __restrict__ basis,
    _Float16* __restrict__ wt, float* __restrict__ cb)
{
    const int idx0   = blockIdx.x * blockDim.x + threadIdx.x;
    const int stride = gridDim.x * blockDim.x;

    for (int idx = idx0; idx < 2 * MLP_HALVES; idx += stride) {
        const int mlp = idx / MLP_HALVES;
        const int r   = idx - mlp * MLP_HALVES;
        float val;
        if (r < 8192) {
            const float* wm;
            int rr;
            if (r < 4096) { wm = mlp ? w1_b : w1_a; rr = r; }
            else          { wm = mlp ? w2_b : w2_a; rr = r - 4096; }
            const int t = rr >> 9, w = rr & 511, lane = w >> 4, h = w & 15;
            const int mt = t >> 1, kt = t & 1;
            const int row = mt * 16 + (lane & 15);
            const int col = kt * 32 + kkA(lane, h);
            val = wm[row * 64 + col];
        } else {
            const int rr = r - 8192;
            const int t = rr >> 9, w = rr & 511, lane = w >> 4, h = w & 15;
            const int mt = t >> 1, kt = t & 1;
            const int i = mt * 16 + (lane & 15);      // coef index 0..31
            const int k = kt * 32 + kkA(lane, h);     // hidden index 0..63
            const float* wo = mlp ? wout_b : wout_a;
            float s = 0.f;
            for (int d = 0; d < 15; ++d) s += basis[i * 15 + d] * wo[d * 64 + k];
            val = s;
        }
        wt[idx] = (_Float16)val;
    }
    for (int idx = idx0; idx < 64; idx += stride) {
        const int mlp = idx >> 5, i = idx & 31;
        const float* bo = mlp ? bout_b : bout_a;
        float s = 0.f;
        for (int d = 0; d < 15; ++d) s += basis[i * 15 + d] * bo[d];
        cb[idx] = s;
    }
}

// ---------------------------------------------------------------------------
// D-layout accumulators (rows = hidden, cols = points) -> next-layer B tile.
// Own half stays in-lane; complementary 8 rows swap with lane^16 via
// ds_swizzle SWAPX16 (packed as 2xf16 per dword).
// e = acc[2*kt], o = acc[2*kt+1]
// ---------------------------------------------------------------------------
__device__ __forceinline__ v16h dconv(const v8f& e, const v8f& o, int kt,
                                      const float* __restrict__ bias,
                                      int hi, bool relu)
{
    const int ownRow = (2 * kt + hi)     * 16 + 8 * hi;
    const int othRow = (2 * kt + 1 - hi) * 16 + 8 * hi;

    unsigned recv[4];
#pragma unroll
    for (int q = 0; q < 4; ++q) {
        float g0 = (hi ? e[2 * q]     : o[2 * q])     + bias[othRow + 2 * q];
        float g1 = (hi ? e[2 * q + 1] : o[2 * q + 1]) + bias[othRow + 2 * q + 1];
        if (relu) { g0 = fmaxf(g0, 0.f); g1 = fmaxf(g1, 0.f); }
        H2 p; p.h[0] = (_Float16)g0; p.h[1] = (_Float16)g1;
        recv[q] = (unsigned)__builtin_amdgcn_ds_swizzle((int)p.u, 0x401f); // SWAPX16
    }

    v16h outv;
#pragma unroll
    for (int r = 0; r < 8; ++r) {
        float f = (hi ? o[r] : e[r]) + bias[ownRow + r];
        if (relu) f = fmaxf(f, 0.f);
        _Float16 own = (_Float16)f;
        H2 p; p.u = recv[r >> 1];
        _Float16 rcv = p.h[r & 1];
        outv[r]     = hi ? rcv : own;
        outv[8 + r] = hi ? own : rcv;
    }
    return outv;
}

// ---------------------------------------------------------------------------
// Full MLP (layer0 scalar, two 64x64 WMMA layers, fused 64->32 output layer).
// Writes 32 coefficients for point (lane&15) of this group into LDS.
// ---------------------------------------------------------------------------
__device__ __forceinline__ void run_mlp(
    float s, const _Float16* __restrict__ wt, const float* __restrict__ cb,
    const float* __restrict__ w0, const float* __restrict__ b0,
    const float* __restrict__ b1, const float* __restrict__ b2,
    float* __restrict__ coefOut, int lane)
{
    const int hi = lane >> 4;

    // layer 0 (1 -> 64), no ReLU, directly in B layout (K = 16*hi + h)
    v16h bt0, bt1;
#pragma unroll
    for (int h = 0; h < 16; ++h) {
        const int j0 = 16 * hi + h;
        const int j1 = 32 + 16 * hi + h;
        bt0[h] = (_Float16)fmaf(s, w0[j0], b0[j0]);
        bt1[h] = (_Float16)fmaf(s, w0[j1], b0[j1]);
    }

    const _Float16* w1t = wt;
    const _Float16* w2t = wt + 4096;
    const _Float16* cwt = wt + 8192;
    v8f acc[4];

    // layer 1 (64 -> 64)
#pragma unroll
    for (int mt = 0; mt < 4; ++mt) {
        v16h a0 = *(const v16h*)(w1t + (mt * 2 + 0) * 512 + lane * 16);
        v16h a1 = *(const v16h*)(w1t + (mt * 2 + 1) * 512 + lane * 16);
        v8f c = {0.f, 0.f, 0.f, 0.f, 0.f, 0.f, 0.f, 0.f};
        c = __builtin_amdgcn_wmma_f32_16x16x32_f16(false, a0, false, bt0, (short)0, c, false, false);
        c = __builtin_amdgcn_wmma_f32_16x16x32_f16(false, a1, false, bt1, (short)0, c, false, false);
        acc[mt] = c;
    }
    bt0 = dconv(acc[0], acc[1], 0, b1, hi, true);
    bt1 = dconv(acc[2], acc[3], 1, b1, hi, true);

    // layer 2 (64 -> 64)
#pragma unroll
    for (int mt = 0; mt < 4; ++mt) {
        v16h a0 = *(const v16h*)(w2t + (mt * 2 + 0) * 512 + lane * 16);
        v16h a1 = *(const v16h*)(w2t + (mt * 2 + 1) * 512 + lane * 16);
        v8f c = {0.f, 0.f, 0.f, 0.f, 0.f, 0.f, 0.f, 0.f};
        c = __builtin_amdgcn_wmma_f32_16x16x32_f16(false, a0, false, bt0, (short)0, c, false, false);
        c = __builtin_amdgcn_wmma_f32_16x16x32_f16(false, a1, false, bt1, (short)0, c, false, false);
        acc[mt] = c;
    }
    bt0 = dconv(acc[0], acc[1], 0, b2, hi, true);
    bt1 = dconv(acc[2], acc[3], 1, b2, hi, true);

    // fused output layer (64 -> 32): theta @ basis^T folded into Cw/Cb
#pragma unroll
    for (int mt = 0; mt < 2; ++mt) {
        v16h a0 = *(const v16h*)(cwt + (mt * 2 + 0) * 512 + lane * 16);
        v16h a1 = *(const v16h*)(cwt + (mt * 2 + 1) * 512 + lane * 16);
        v8f c = {0.f, 0.f, 0.f, 0.f, 0.f, 0.f, 0.f, 0.f};
        c = __builtin_amdgcn_wmma_f32_16x16x32_f16(false, a0, false, bt0, (short)0, c, false, false);
        c = __builtin_amdgcn_wmma_f32_16x16x32_f16(false, a1, false, bt1, (short)0, c, false, false);
        const int i0 = mt * 16 + 8 * hi;
#pragma unroll
        for (int r = 0; r < 8; ++r) coefOut[i0 + r] = c[r] + cb[i0 + r];
    }
}

// ---------------------------------------------------------------------------
// CPAB cell walk (17 fixed iterations, matches jax fori_loop semantics)
// ---------------------------------------------------------------------------
__device__ __forceinline__ void cpab_solve(float x0, const float* __restrict__ cf,
                                           float& resOut, float& logaOut)
{
    float xx = x0, t = 1.f, loga = 0.f, res = x0;
    int c = (int)(x0 * 16.f);
    c = max(0, min(15, c));
    bool done = false;
#pragma unroll 1
    for (int it = 0; it < 17; ++it) {
        const float a = cf[2 * c];
        const float b = cf[2 * c + 1];
        const float left  = (float)c * 0.0625f;
        const float right = (float)(c + 1) * 0.0625f;
        const bool  small = fabsf(a) < 1e-8f;
        const float asafe = small ? 1.f : a;
        const float xb    = xx + b / asafe;
        const float psi   = small ? (xx + t * b) : (__expf(t * a) * xb - b / asafe);
        const bool  inside = (psi >= left) && (psi <= right);
        const float vv    = a * xx + b;
        const float xc    = (vv >= 0.f) ? right : left;
        const float bsafe  = (fabsf(b)  < 1e-12f) ? 1.f : b;
        const float xbsafe = (fabsf(xb) < 1e-12f) ? 1.f : xb;
        const float ratio  = (xc + b / asafe) / xbsafe;
        const float thit   = small ? ((xc - xx) / bsafe)
                                   : (__logf(fmaxf(ratio, 1e-12f)) / asafe);
        const float dt = inside ? t : thit;
        loga += done ? 0.f : a * dt;
        res = done ? res : (inside ? psi : res);
        const bool newdone = done | inside;
        xx = newdone ? xx : xc;
        t  = newdone ? t : (t - thit);
        int cn = c + ((vv >= 0.f) ? 1 : -1);
        cn = max(0, min(15, cn));
        c = newdone ? c : cn;
        done = newdone;
    }
    resOut = res;
    logaOut = loga;
}

// ---------------------------------------------------------------------------
// Fused main kernel: 256 points / block, 32 points / wave.
// ---------------------------------------------------------------------------
__global__ __launch_bounds__(256) void cpab_fused(
    const float* __restrict__ x,
    const _Float16* __restrict__ wt, const float* __restrict__ cbAll,
    const float* __restrict__ w0_2, const float* __restrict__ b0_2,
    const float* __restrict__ b1_2, const float* __restrict__ b2_2,
    const float* __restrict__ w0_1, const float* __restrict__ b0_1,
    const float* __restrict__ b1_1, const float* __restrict__ b2_1,
    float* __restrict__ out, int Bn)
{
    __shared__ float s_coef[256 * COEF_STR];
    __shared__ float s_z[256];

    const int tid  = threadIdx.x;
    const int lane = tid & 31;
    const int wave = tid >> 5;
    const int m    = lane & 15;
    const int wbase = blockIdx.x * 256 + wave * 32;

    const int pl  = wbase + lane;
    const int plc = min(pl, Bn - 1);
    const float u = fminf(fmaxf(x[2 * plc + 1], EPS_CLIP), 1.f - EPS_CLIP); // x1
    const float v = fminf(fmaxf(x[2 * plc + 0], EPS_CLIP), 1.f - EPS_CLIP); // x2

    // ---- theta2 = MLP2(x1) ----
#pragma unroll
    for (int g = 0; g < 2; ++g) {
        const int pg = min(wbase + g * 16 + m, Bn - 1);
        const float s = fminf(fmaxf(x[2 * pg + 1], EPS_CLIP), 1.f - EPS_CLIP);
        run_mlp(s, wt, cbAll, w0_2, b0_2, b1_2, b2_2,
                &s_coef[(wave * 32 + g * 16 + m) * COEF_STR], lane);
    }
    __syncthreads();

    float z2, lg2;
    cpab_solve(v, &s_coef[(wave * 32 + lane) * COEF_STR], z2, lg2);
    __syncthreads();
    s_z[tid] = z2;
    __syncthreads();

    // ---- theta1 = MLP1(z2) ----
#pragma unroll
    for (int g = 0; g < 2; ++g) {
        const float s = s_z[wave * 32 + g * 16 + m];
        run_mlp(s, wt + MLP_HALVES, cbAll + 32, w0_1, b0_1, b1_1, b2_1,
                &s_coef[(wave * 32 + g * 16 + m) * COEF_STR], lane);
    }
    __syncthreads();

    float z1, lg1;
    cpab_solve(u, &s_coef[(wave * 32 + lane) * COEF_STR], z1, lg1);

    if (pl < Bn) {
        out[2 * pl + 0] = z2;   // z[:, mask] column 0 = z2
        out[2 * pl + 1] = z1;
        const long lo = 2L * (long)Bn;
        out[lo + 2 * pl + 0] = lg2;
        out[lo + 2 * pl + 1] = lg1;
    }
}

// ---------------------------------------------------------------------------
extern "C" void kernel_launch(void* const* d_in, const int* in_sizes, int n_in,
                              void* d_out, int out_size, void* d_ws, size_t ws_size,
                              hipStream_t stream)
{
    // Detect top-level flattening order: dict-order (x first) vs jax alphabetical (basis first)
    int xi, p1, p2, bi;
    if (n_in >= 18 && in_sizes[0] == 480) { bi = 0; p1 = 1; p2 = 9; xi = 17; }
    else                                  { xi = 0; p1 = 1; p2 = 9; bi = 17; }

    // Detect per-params ordering: dict (w0,b0,w1,b1,w2,b2,wout,bout) vs alphabetical
    int ow0, ob0, ow1, ob1, ow2, ob2, owo, obo;
    if (in_sizes[p1 + 2] == 4096) { ow0 = 0; ob0 = 1; ow1 = 2; ob1 = 3; ow2 = 4; ob2 = 5; owo = 6; obo = 7; }
    else                          { ob0 = 0; ob1 = 1; ob2 = 2; obo = 3; ow0 = 4; ow1 = 5; ow2 = 6; owo = 7; }

    const float* x     = (const float*)d_in[xi];
    const float* basis = (const float*)d_in[bi];
    auto F = [&](int g, int o) { return (const float*)d_in[g + o]; };

    _Float16* wt = (_Float16*)d_ws;
    float*    cb = (float*)((char*)d_ws + (size_t)(2 * MLP_HALVES) * sizeof(_Float16));
    const int Bn = in_sizes[xi] / 2;

    prep_weights<<<40, 256, 0, stream>>>(
        F(p2, ow1), F(p2, ow2), F(p2, owo), F(p2, obo),   // mlp slot 0 = params2
        F(p1, ow1), F(p1, ow2), F(p1, owo), F(p1, obo),   // mlp slot 1 = params1
        basis, wt, cb);

    cpab_fused<<<(Bn + 255) / 256, 256, 0, stream>>>(
        x, wt, cb,
        F(p2, ow0), F(p2, ob0), F(p2, ob1), F(p2, ob2),
        F(p1, ow0), F(p1, ob0), F(p1, ob1), F(p1, ob2),
        (float*)d_out, Bn);
}
